// LlamaCrossAttention_4363686772886
// MI455X (gfx1250) — matrix operand order repaired and checked by reading, not compiled, not run on hardware
//
#include <hip/hip_runtime.h>
#include <hip/hip_bf16.h>

#define B_    2
#define Q_    1024
#define HID_  2048
#define LH_   32
#define LD_   128
#define KV_   2048
#define LHID_ 4096
#define BQ_   (B_*Q_)

typedef __bf16 bf16_t;
typedef __attribute__((ext_vector_type(8)))  __bf16 v8bf;
typedef __attribute__((ext_vector_type(16))) __bf16 v16bf;
typedef __attribute__((ext_vector_type(8)))  float  v8f;

// -log2(10000)/64 : inv_freq[i] = 10000^(-i/64) = exp2(i * kRopeCoef)
#define kRopeCoef (-0.20762050593045952f)

// A-matrix fragment (16x32 bf16, MxK). Lane: m = lane&15, h = lane>>4.
// Per-lane elements: e=0..15 -> K = 16h+e.
// Caller passes p = row_base + k0 + 16*h  (row-major, K contiguous).
__device__ __forceinline__ v16bf load_frag_a(const bf16_t* p) {
  v8bf lo = *reinterpret_cast<const v8bf*>(p);
  v8bf hi = *reinterpret_cast<const v8bf*>(p + 8);
  return __builtin_shufflevector(lo, hi, 0,1,2,3,4,5,6,7,8,9,10,11,12,13,14,15);
}

// B-matrix fragment (32x16 bf16, KxN) loaded from B^T stored N-major/K-contiguous.
// Lane: n = lane&15, h = lane>>4. Elements e=0..15 -> K = 16h + e.
// Caller passes p = row_base(n) + k0 + 16*h.
__device__ __forceinline__ v16bf load_frag_b(const bf16_t* p) {
  v8bf lo = *reinterpret_cast<const v8bf*>(p);
  v8bf hi = *reinterpret_cast<const v8bf*>(p + 8);
  return __builtin_shufflevector(lo, hi, 0,1,2,3,4,5,6,7,8,9,10,11,12,13,14,15);
}

__device__ __forceinline__ v8f wmma_bf16(v16bf a, v16bf b, v8f c) {
  return __builtin_amdgcn_wmma_f32_16x16x32_bf16(false, a, false, b, (short)0, c,
                                                 false, false);
}

// ---------------------------------------------------------------- fp32 -> bf16
__global__ void cvt_f32_bf16(const float* __restrict__ src,
                             bf16_t* __restrict__ dst, long n) {
  long i = (long)blockIdx.x * blockDim.x + threadIdx.x;
  long stride = (long)gridDim.x * blockDim.x;
  for (; i < n; i += stride) dst[i] = (bf16_t)src[i];
}

// ------------------------------------------- q = hidden @ Wq^T, +RoPE(pos_ids)
// One wave per (16-row tile of BQ, head). Wave computes 16x128 (full head dim),
// so rotate_half (d <-> d+-64) stays inside the wave's 8 accumulators.
__global__ __launch_bounds__(128) void qproj_rope_kernel(
    const bf16_t* __restrict__ hb, const bf16_t* __restrict__ wq,
    const int* __restrict__ pos, bf16_t* __restrict__ qbf) {
  const int wave = blockIdx.x * 4 + (threadIdx.x >> 5);
  const int lane = threadIdx.x & 31;
  const int mloc = lane & 15, hl = lane >> 4;
  const int head = wave & (LH_ - 1);
  const int row0 = (wave >> 5) * 16;          // global row in [0, BQ)
  const int b = row0 / Q_;
  const int q0 = row0 - b * Q_;

  v8f acc[8] = {};
  const bf16_t* pa = hb + (size_t)(row0 + mloc) * HID_ + 16 * hl;
  const bf16_t* pb = wq + ((size_t)head * LD_ + mloc) * HID_ + 16 * hl;

  for (int kk = 0; kk < HID_; kk += 32) {
    __builtin_prefetch(pa + kk + 256, 0, 1);
    v16bf a = load_frag_a(pa + kk);
#pragma unroll
    for (int t = 0; t < 8; ++t)
      acc[t] = wmma_bf16(a, load_frag_b(pb + (size_t)(t * 16) * HID_ + kk), acc[t]);
  }
  // RoPE epilogue: d = t*16 + mloc (t<4 -> d<64), pair (d, d+64) shares angle.
  float invf[4];
#pragma unroll
  for (int t = 0; t < 4; ++t)
    invf[t] = exp2f((float)(t * 16 + mloc) * kRopeCoef);
#pragma unroll
  for (int r = 0; r < 8; ++r) {
    const int m2 = r + 8 * hl;
    const float p = (float)pos[b * Q_ + q0 + m2];
    bf16_t* orow = qbf + ((size_t)(b * LH_ + head) * Q_ + q0 + m2) * LD_;
#pragma unroll
    for (int t = 0; t < 4; ++t) {
      const int d = t * 16 + mloc;
      const float ang = p * invf[t];
      const float cv = __cosf(ang), sv = __sinf(ang);
      const float xlo = acc[t][r];          // element d
      const float xhi = acc[t + 4][r];      // element d+64
      orow[d]      = (bf16_t)(xlo * cv - xhi * sv);
      orow[d + 64] = (bf16_t)(xhi * cv + xlo * sv);
    }
  }
}

// --------------------------------- k = large_k @ Wk^T + bk, +RoPE(arange(KV))
__global__ __launch_bounds__(128) void kremap_rope_kernel(
    const bf16_t* __restrict__ lk, const bf16_t* __restrict__ wk,
    const float* __restrict__ bk, bf16_t* __restrict__ kbf) {
  const int wave = blockIdx.x * 4 + (threadIdx.x >> 5);
  const int lane = threadIdx.x & 31;
  const int mloc = lane & 15, hl = lane >> 4;
  const int kvTile = wave & 127;
  const int h = (wave >> 7) & (LH_ - 1);
  const int b = wave >> 12;
  const int row0 = kvTile * 16;

  v8f acc[8] = {};
  const bf16_t* pa = lk + ((size_t)(b * LH_ + h) * KV_ + row0 + mloc) * LD_ + 16 * hl;
  const bf16_t* pb = wk + (size_t)mloc * LD_ + 16 * hl;
#pragma unroll
  for (int kk = 0; kk < LD_; kk += 32) {
    v16bf a = load_frag_a(pa + kk);
#pragma unroll
    for (int t = 0; t < 8; ++t)
      acc[t] = wmma_bf16(a, load_frag_b(pb + (size_t)(t * 16) * LD_ + kk), acc[t]);
  }
  float invf[4], bkl[4], bkh[4];
#pragma unroll
  for (int t = 0; t < 4; ++t) {
    const int d = t * 16 + mloc;
    invf[t] = exp2f((float)d * kRopeCoef);
    bkl[t] = bk[d];
    bkh[t] = bk[d + 64];
  }
#pragma unroll
  for (int r = 0; r < 8; ++r) {
    const int m2 = r + 8 * hl;
    const float p = (float)(row0 + m2);
    bf16_t* orow = kbf + ((size_t)(b * LH_ + h) * KV_ + row0 + m2) * LD_;
#pragma unroll
    for (int t = 0; t < 4; ++t) {
      const int d = t * 16 + mloc;
      const float ang = p * invf[t];
      const float cv = __cosf(ang), sv = __sinf(ang);
      const float xlo = acc[t][r] + bkl[t];
      const float xhi = acc[t + 4][r] + bkh[t];
      orow[d]      = (bf16_t)(xlo * cv - xhi * sv);
      orow[d + 64] = (bf16_t)(xhi * cv + xlo * sv);
    }
  }
}

// ------------------- v = large_v @ Wv^T + bv, stored transposed v_t[b,h,d,kv]
__global__ __launch_bounds__(128) void vremap_kernel(
    const bf16_t* __restrict__ lv, const bf16_t* __restrict__ wv,
    const float* __restrict__ bv, bf16_t* __restrict__ vt) {
  const int wave = blockIdx.x * 4 + (threadIdx.x >> 5);
  const int lane = threadIdx.x & 31;
  const int mloc = lane & 15, hl = lane >> 4;
  const int kvTile = wave & 127;
  const int h = (wave >> 7) & (LH_ - 1);
  const int b = wave >> 12;
  const int row0 = kvTile * 16;

  v8f acc[8] = {};
  const bf16_t* pa = lv + ((size_t)(b * LH_ + h) * KV_ + row0 + mloc) * LD_ + 16 * hl;
  const bf16_t* pb = wv + (size_t)mloc * LD_ + 16 * hl;
#pragma unroll
  for (int kk = 0; kk < LD_; kk += 32) {
    v16bf a = load_frag_a(pa + kk);
#pragma unroll
    for (int t = 0; t < 8; ++t)
      acc[t] = wmma_bf16(a, load_frag_b(pb + (size_t)(t * 16) * LD_ + kk), acc[t]);
  }
  float bvv[8];
#pragma unroll
  for (int t = 0; t < 8; ++t) bvv[t] = bv[t * 16 + mloc];
  bf16_t* base = vt + (size_t)(b * LH_ + h) * LD_ * KV_;
#pragma unroll
  for (int r = 0; r < 8; ++r) {
    const int m2 = r + 8 * hl;
#pragma unroll
    for (int t = 0; t < 8; ++t) {
      const int d = t * 16 + mloc;
      base[(size_t)d * KV_ + row0 + m2] = (bf16_t)(acc[t][r] + bvv[t]);
    }
  }
}

// ------------------------------------------------ flash attention, wave-level
// One wave per (b, h, 16-row q tile); streams KV in 32-steps.
__global__ __launch_bounds__(128) void flash_attn_kernel(
    const bf16_t* __restrict__ qbf, const bf16_t* __restrict__ kbf,
    const bf16_t* __restrict__ vt, const float* __restrict__ mask,
    bf16_t* __restrict__ ao) {
  __shared__ __align__(16) bf16_t ldsP[4][16 * 32];
  const int ws = threadIdx.x >> 5;
  const int wave = blockIdx.x * 4 + ws;
  const int lane = threadIdx.x & 31;
  const int mloc = lane & 15, hl = lane >> 4;
  const int qTile = wave & 63;
  const int h = (wave >> 6) & (LH_ - 1);
  const int b = wave >> 11;
  const int q0 = qTile * 16;

  const bf16_t* pq = qbf + ((size_t)(b * LH_ + h) * Q_ + q0 + mloc) * LD_ + 16 * hl;
  v16bf aq[4];
#pragma unroll
  for (int j = 0; j < 4; ++j) aq[j] = load_frag_a(pq + j * 32);

  v8f acc[8] = {};
  float mi[8], li[8];
#pragma unroll
  for (int r = 0; r < 8; ++r) { mi[r] = -3.0e38f; li[r] = 0.0f; }

  const float scale = 0.08838834764831845f;  // 1/sqrt(128)
  const float* mrow = mask + ((size_t)b * Q_ + q0) * KV_;
  const bf16_t* kb = kbf + (size_t)(b * LH_ + h) * KV_ * LD_;
  const bf16_t* vb = vt + (size_t)(b * LH_ + h) * LD_ * KV_;
  bf16_t* pl = &ldsP[ws][0];

  for (int kv0 = 0; kv0 < KV_; kv0 += 32) {
    v8f s0 = {}, s1 = {};
    {
      const bf16_t* pk = kb + (size_t)(kv0 + mloc) * LD_ + 16 * hl;
#pragma unroll
      for (int j = 0; j < 4; ++j) s0 = wmma_bf16(aq[j], load_frag_b(pk + j * 32), s0);
    }
    {
      const bf16_t* pk = kb + (size_t)(kv0 + 16 + mloc) * LD_ + 16 * hl;
#pragma unroll
      for (int j = 0; j < 4; ++j) s1 = wmma_bf16(aq[j], load_frag_b(pk + j * 32), s1);
    }
    float alpha[8];
#pragma unroll
    for (int r = 0; r < 8; ++r) {
      const int m2 = r + 8 * hl;
      const float* mr = mrow + (size_t)m2 * KV_ + kv0 + mloc;
      const float a0 = s0[r] * scale + mr[0];
      const float a1 = s1[r] * scale + mr[16];
      float tmx = fmaxf(a0, a1);
      tmx = fmaxf(tmx, __shfl_xor(tmx, 1, 32));
      tmx = fmaxf(tmx, __shfl_xor(tmx, 2, 32));
      tmx = fmaxf(tmx, __shfl_xor(tmx, 4, 32));
      tmx = fmaxf(tmx, __shfl_xor(tmx, 8, 32));
      const float mnew = fmaxf(mi[r], tmx);
      alpha[r] = __expf(mi[r] - mnew);
      const float p0 = __expf(a0 - mnew);
      const float p1 = __expf(a1 - mnew);
      float rs = p0 + p1;
      rs += __shfl_xor(rs, 1, 32);
      rs += __shfl_xor(rs, 2, 32);
      rs += __shfl_xor(rs, 4, 32);
      rs += __shfl_xor(rs, 8, 32);
      li[r] = li[r] * alpha[r] + rs;
      mi[r] = mnew;
      // stage P (bf16) into per-wave LDS tile [16 q rows][32 kv]
      pl[m2 * 32 + mloc] = (bf16_t)p0;
      pl[m2 * 32 + 16 + mloc] = (bf16_t)p1;
    }
#pragma unroll
    for (int t = 0; t < 8; ++t)
#pragma unroll
      for (int r = 0; r < 8; ++r) acc[t][r] *= alpha[r];
    // re-load P as an A fragment (wave-private LDS, DS ops are in-order)
    v16bf ap = load_frag_a(pl + mloc * 32 + 16 * hl);
#pragma unroll
    for (int t = 0; t < 8; ++t) {
      const bf16_t* pv = vb + (size_t)(t * 16 + mloc) * KV_ + kv0 + 16 * hl;
      acc[t] = wmma_bf16(ap, load_frag_b(pv), acc[t]);
    }
  }
#pragma unroll
  for (int r = 0; r < 8; ++r) {
    const int m2 = r + 8 * hl;
    const float inv = 1.0f / li[r];
    bf16_t* orow = ao + (size_t)(b * Q_ + q0 + m2) * LHID_ + h * LD_;
#pragma unroll
    for (int t = 0; t < 8; ++t)
      orow[t * 16 + mloc] = (bf16_t)(acc[t][r] * inv);
  }
}

// ------------------------------------------------------- out = ao @ Wo^T (f32)
__global__ __launch_bounds__(128) void oproj_kernel(
    const bf16_t* __restrict__ ao, const bf16_t* __restrict__ wo,
    float* __restrict__ out) {
  const int wave = blockIdx.x * 4 + (threadIdx.x >> 5);
  const int lane = threadIdx.x & 31;
  const int mloc = lane & 15, hl = lane >> 4;
  const int colTile = wave & 31;   // 32 tiles of 64 cols
  const int row0 = (wave >> 5) * 16;
  const int n0 = colTile * 64;

  v8f acc[4] = {};
  const bf16_t* pa = ao + (size_t)(row0 + mloc) * LHID_ + 16 * hl;
  const bf16_t* pb = wo + (size_t)(n0 + mloc) * LHID_ + 16 * hl;
  for (int kk = 0; kk < LHID_; kk += 32) {
    __builtin_prefetch(pa + kk + 256, 0, 1);
    v16bf a = load_frag_a(pa + kk);
#pragma unroll
    for (int t = 0; t < 4; ++t)
      acc[t] = wmma_bf16(a, load_frag_b(pb + (size_t)(t * 16) * LHID_ + kk), acc[t]);
  }
#pragma unroll
  for (int r = 0; r < 8; ++r) {
    float* orow = out + (size_t)(row0 + r + 8 * hl) * HID_ + n0;
#pragma unroll
    for (int t = 0; t < 4; ++t) orow[t * 16 + mloc] = acc[t][r];
  }
}

extern "C" void kernel_launch(void* const* d_in, const int* in_sizes, int n_in,
                              void* d_out, int out_size, void* d_ws, size_t ws_size,
                              hipStream_t stream) {
  const float* hidden = (const float*)d_in[0];
  const float* mask   = (const float*)d_in[1];
  const int*   pos    = (const int*)d_in[2];
  const float* lk     = (const float*)d_in[3];
  const float* lv     = (const float*)d_in[4];
  const float* Wq     = (const float*)d_in[5];
  const float* Wo     = (const float*)d_in[6];
  const float* Wk     = (const float*)d_in[7];
  const float* bk     = (const float*)d_in[8];
  const float* Wv     = (const float*)d_in[9];
  const float* bv     = (const float*)d_in[10];
  float* out = (float*)d_out;

  char* w = (char*)d_ws;
  auto alloc = [&](size_t bytes) {
    char* p = w;
    w += (bytes + 255) & ~(size_t)255;
    return p;
  };
  bf16_t* hb  = (bf16_t*)alloc((size_t)BQ_ * HID_ * 2);
  bf16_t* wqb = (bf16_t*)alloc((size_t)LHID_ * HID_ * 2);
  bf16_t* wob = (bf16_t*)alloc((size_t)HID_ * LHID_ * 2);
  bf16_t* wkb = (bf16_t*)alloc((size_t)LD_ * LD_ * 2);
  bf16_t* wvb = (bf16_t*)alloc((size_t)LD_ * LD_ * 2);
  bf16_t* lkb = (bf16_t*)alloc((size_t)B_ * LH_ * KV_ * LD_ * 2);
  bf16_t* lvb = (bf16_t*)alloc((size_t)B_ * LH_ * KV_ * LD_ * 2);
  bf16_t* qb  = (bf16_t*)alloc((size_t)B_ * LH_ * Q_ * LD_ * 2);
  bf16_t* kbf = (bf16_t*)alloc((size_t)B_ * LH_ * KV_ * LD_ * 2);
  bf16_t* vtb = (bf16_t*)alloc((size_t)B_ * LH_ * LD_ * KV_ * 2);
  bf16_t* aob = (bf16_t*)alloc((size_t)BQ_ * LHID_ * 2);

  auto cvt = [&](const float* s, bf16_t* d, long n) {
    cvt_f32_bf16<<<2048, 256, 0, stream>>>(s, d, n);
  };
  cvt(hidden, hb, (long)BQ_ * HID_);
  cvt(Wq, wqb, (long)LHID_ * HID_);
  cvt(Wo, wob, (long)HID_ * LHID_);
  cvt(Wk, wkb, (long)LD_ * LD_);
  cvt(Wv, wvb, (long)LD_ * LD_);
  cvt(lk, lkb, (long)B_ * LH_ * KV_ * LD_);
  cvt(lv, lvb, (long)B_ * LH_ * KV_ * LD_);

  qproj_rope_kernel<<<1024, 128, 0, stream>>>(hb, wqb, pos, qb);   // 4096 waves
  kremap_rope_kernel<<<2048, 128, 0, stream>>>(lkb, wkb, bk, kbf); // 8192 waves
  vremap_kernel<<<2048, 128, 0, stream>>>(lvb, wvb, bv, vtb);      // 8192 waves
  flash_attn_kernel<<<1024, 128, 0, stream>>>(qb, kbf, vtb, mask, aob);
  oproj_kernel<<<1024, 128, 0, stream>>>(aob, wob, out);           // 4096 waves
}
